// ReprogrammingLayerLin_34299608826433
// MI455X (gfx1250) — compile-verified, matrix operand order checked by reading, not run on hardware
//
#include <hip/hip_runtime.h>
#include <hip/hip_bf16.h>
#include <stdint.h>

// ---------------------------------------------------------------------------
// ReprogrammingLayer (cross attention, ELU+L2-normalized Q/K) for MI455X
// gfx1250: GEMMs via v_wmma_f32_16x16x32_bf16, staging via
// global_load_async_to_lds_b128 (ASYNCcnt), split barriers, prefetch.
// ---------------------------------------------------------------------------

typedef __attribute__((ext_vector_type(16))) __bf16 bf16x16;
typedef __attribute__((ext_vector_type(8)))  float  f32x8;

#define D_MODEL 1024
#define SEQ_S   2048
#define ROWS_Q  4096   // B*L
#define NHEADS  16
#define EHEAD   64

__device__ __forceinline__ unsigned short f2bf(float f) {
    union { float f; unsigned u; } v; v.f = f;
    unsigned r = v.u + 0x7fffu + ((v.u >> 16) & 1u);   // round-to-nearest-even
    return (unsigned short)(r >> 16);
}

// CDNA5 async copy: 16B global -> LDS, tracked by ASYNCcnt (no VGPR data).
__device__ __forceinline__ void async_cp_b128(void* lds, const void* g) {
    unsigned           l = (unsigned)(uintptr_t)lds;
    unsigned long long a = (unsigned long long)(uintptr_t)g;
    asm volatile("global_load_async_to_lds_b128 %0, %1, off"
                 :: "v"(l), "v"(a) : "memory");
}
__device__ __forceinline__ void wait_async0() {
    asm volatile("s_wait_asynccnt 0x0" ::: "memory");
}

// ---------------------------------------------------------------------------
// fp32 -> bf16 elementwise cast (4/thread)
// ---------------------------------------------------------------------------
__global__ __launch_bounds__(256)
void cast_bf16_kernel(const float* __restrict__ in, unsigned short* __restrict__ out, int n) {
    int i = (blockIdx.x * 256 + threadIdx.x) * 4;
    if (i >= n) return;
    float4 v = *(const float4*)(in + i);
    uint2 p;
    p.x = (unsigned)f2bf(v.x) | ((unsigned)f2bf(v.y) << 16);
    p.y = (unsigned)f2bf(v.z) | ((unsigned)f2bf(v.w) << 16);
    *(uint2*)(out + i) = p;
}

// ---------------------------------------------------------------------------
// W (K x N, fp32) -> WT (N x K, bf16). 32x32 LDS tiles.
// ---------------------------------------------------------------------------
__global__ __launch_bounds__(256)
void transpose_cast_kernel(const float* __restrict__ W, unsigned short* __restrict__ WT) {
    __shared__ float t[32][33];
    const int n0 = blockIdx.x * 32, k0 = blockIdx.y * 32;
    const int tx = threadIdx.x, ty = threadIdx.y;       // block (32,8)
#pragma unroll
    for (int i = 0; i < 4; ++i)
        t[ty + 8 * i][tx] = W[(size_t)(k0 + ty + 8 * i) * D_MODEL + n0 + tx];
    __syncthreads();
#pragma unroll
    for (int i = 0; i < 4; ++i)
        WT[(size_t)(n0 + ty + 8 * i) * D_MODEL + k0 + tx] = f2bf(t[tx][ty + 8 * i]);
}

// ---------------------------------------------------------------------------
// Tiled WMMA GEMM: C = A(MxK) * BT(NxK)^T, K = N = 1024.
// 256 threads (8 waves), tile 128x128, K-step 32, double-buffered async-to-LDS.
// modes: 0=Q (bias+elu+head L2 norm, row-major), 1=K (same, [h][s][e]),
//        2=V (bias, [h][e][s]), 3=O (bias+residual, fp32 row-major)
// ---------------------------------------------------------------------------
#define SMLD 40   // padded LDS row stride (halves); 80B rows keep 16B alignment

__global__ __launch_bounds__(256)
void gemm_wmma_kernel(const unsigned short* __restrict__ A,
                      const unsigned short* __restrict__ BT,
                      const float* __restrict__ bias,
                      const float* __restrict__ resid,
                      void* __restrict__ out, int M, int mode) {
    __shared__ unsigned short smA[2][128 * SMLD];
    __shared__ unsigned short smB[2][128 * SMLD];

    const int tid  = threadIdx.x;
    const int lane = tid & 31, wid = tid >> 5;
    const int hl   = lane >> 4, lid = lane & 15;
    const int wm   = wid & 3,  wn  = wid >> 2;
    const int bm   = blockIdx.x * 128, bn = blockIdx.y * 128;

    f32x8 acc[2][4];
#pragma unroll
    for (int i = 0; i < 2; ++i)
#pragma unroll
        for (int j = 0; j < 4; ++j)
#pragma unroll
            for (int r = 0; r < 8; ++r) acc[i][j][r] = 0.f;

    // staging geometry: tile = 128 rows x 32 halves = 512 x 16B chunks,
    // 2 chunks per thread per operand.
    const int c0   = tid * 2;
    const int srow = c0 >> 2;              // 0..127
    const int scol = (c0 & 3) * 8;         // halves
    const unsigned short* gA = A  + (size_t)(bm + srow) * D_MODEL + scol;
    const unsigned short* gB = BT + (size_t)(bn + srow) * D_MODEL + scol;

    auto stage = [&](int buf, int k0) {
#pragma unroll
        for (int c = 0; c < 2; ++c) {
            async_cp_b128(&smA[buf][srow * SMLD + scol + 8 * c], gA + k0 + 8 * c);
            async_cp_b128(&smB[buf][srow * SMLD + scol + 8 * c], gB + k0 + 8 * c);
        }
    };

    stage(0, 0);                                   // prologue: tile 0 in flight
    const int niter = D_MODEL / 32;
    for (int it = 0; it < niter; ++it) {
        const int cur = it & 1;
        wait_async0();                             // this wave's copies of tile it
        __syncthreads();                           // everyone's copies done
        if (it + 1 < niter) {
            stage(cur ^ 1, (it + 1) * 32);         // copy tile it+1 under compute
            if (it + 2 < niter) {                  // warm L2 two tiles ahead
                __builtin_prefetch(gA + (it + 2) * 32, 0, 1);
                __builtin_prefetch(gB + (it + 2) * 32, 0, 1);
            }
        }

        bf16x16 af[2], bfm[4];
#pragma unroll
        for (int i = 0; i < 2; ++i) {
            const unsigned short* p = &smA[cur][(wm * 32 + i * 16 + lid) * SMLD + hl * 8];
            ((uint4*)&af[i])[0] = *(const uint4*)p;
            ((uint4*)&af[i])[1] = *(const uint4*)(p + 16);
        }
#pragma unroll
        for (int j = 0; j < 4; ++j) {
            const unsigned short* p = &smB[cur][(wn * 64 + j * 16 + lid) * SMLD + hl * 8];
            ((uint4*)&bfm[j])[0] = *(const uint4*)p;
            ((uint4*)&bfm[j])[1] = *(const uint4*)(p + 16);
        }
#pragma unroll
        for (int i = 0; i < 2; ++i)
#pragma unroll
            for (int j = 0; j < 4; ++j)
                acc[i][j] = __builtin_amdgcn_wmma_f32_16x16x32_bf16(
                    false, af[i], false, bfm[j], (short)0, acc[i][j], false, false);
    }

    // ---------------- epilogue ----------------
#pragma unroll
    for (int i = 0; i < 2; ++i) {
        const int gmBase = bm + wm * 32 + i * 16;
        if (mode <= 1) {                                  // bias + elu + head L2 norm
            float ss[8];
#pragma unroll
            for (int r = 0; r < 8; ++r) ss[r] = 0.f;
#pragma unroll
            for (int j = 0; j < 4; ++j) {
                const int gn = bn + wn * 64 + j * 16 + lid;
                const float bv = bias[gn];
#pragma unroll
                for (int r = 0; r < 8; ++r) {
                    float x = acc[i][j][r] + bv;
                    float t = x > 0.f ? x : (__expf(x) - 1.f);
                    acc[i][j][r] = t;
                    ss[r] += t * t;
                }
            }
#pragma unroll
            for (int r = 0; r < 8; ++r) {
                float s = ss[r];
                s += __shfl_xor(s, 1, 16); s += __shfl_xor(s, 2, 16);
                s += __shfl_xor(s, 4, 16); s += __shfl_xor(s, 8, 16);
                ss[r] = rsqrtf(s);                        // 1 / ||head vec||
            }
            unsigned short* outb = (unsigned short*)out;
#pragma unroll
            for (int j = 0; j < 4; ++j) {
                const int gn = bn + wn * 64 + j * 16 + lid;
#pragma unroll
                for (int r = 0; r < 8; ++r) {
                    const int gm = gmBase + r + 8 * hl;
                    unsigned short v = f2bf(acc[i][j][r] * ss[r]);
                    if (mode == 0) {
                        outb[(size_t)gm * D_MODEL + gn] = v;          // qn row-major
                    } else {
                        const int h = gn >> 6, e = gn & 63;
                        outb[((size_t)h * SEQ_S + gm) * EHEAD + e] = v; // kn [h][s][e]
                    }
                }
            }
        } else if (mode == 2) {                           // V: bias only, [h][e][s]
            unsigned short* outb = (unsigned short*)out;
#pragma unroll
            for (int j = 0; j < 4; ++j) {
                const int gn = bn + wn * 64 + j * 16 + lid;
                const float bv = bias[gn];
                const int h = gn >> 6, e = gn & 63;
#pragma unroll
                for (int r = 0; r < 8; ++r) {
                    const int gm = gmBase + r + 8 * hl;
                    outb[((size_t)h * EHEAD + e) * SEQ_S + gm] = f2bf(acc[i][j][r] + bv);
                }
            }
        } else {                                          // O: bias + residual, fp32
            float* outf = (float*)out;
#pragma unroll
            for (int j = 0; j < 4; ++j) {
                const int gn = bn + wn * 64 + j * 16 + lid;
                const float bv = bias[gn];
#pragma unroll
                for (int r = 0; r < 8; ++r) {
                    const int gm = gmBase + r + 8 * hl;
                    const size_t idx = (size_t)gm * D_MODEL + gn;
                    outf[idx] = acc[i][j][r] + bv + resid[idx];
                }
            }
        }
    }
}

// ---------------------------------------------------------------------------
// Flash-style attention: grid (L/64, H, B), 128 threads (4 waves x 16 L rows).
// K/V tiles staged with async-to-LDS; scores & ctx via WMMA; online softmax.
// ---------------------------------------------------------------------------
#define KLD 72   // padded LDS stride (halves), 144B rows keep 16B alignment

__global__ __launch_bounds__(128)
void attn_wmma_kernel(const unsigned short* __restrict__ qn,   // (B*L) x 1024
                      const unsigned short* __restrict__ kn,   // [h][s][e]
                      const unsigned short* __restrict__ vT,   // [h][e][s]
                      unsigned short* __restrict__ ctx) {      // (B*L) x 1024
    __shared__ unsigned short smK[64 * KLD];
    __shared__ unsigned short smV[64 * KLD];
    __shared__ unsigned short smP[4 * 16 * KLD];

    const int tid  = threadIdx.x;
    const int lane = tid & 31, wid = tid >> 5;
    const int hl   = lane >> 4, lid = lane & 15;
    const int b = blockIdx.z, h = blockIdx.y;
    const int lrow = blockIdx.x * 64 + wid * 16;
    const float scale = 0.125f;                        // 1/sqrt(64)

    // Q fragments held in registers for the whole S loop
    bf16x16 aq[2];
    {
        const unsigned short* qp =
            qn + (size_t)(b * SEQ_S + lrow + lid) * D_MODEL + h * EHEAD;
#pragma unroll
        for (int kk = 0; kk < 2; ++kk) {
            const unsigned short* p = qp + kk * 32 + hl * 8;
            ((uint4*)&aq[kk])[0] = *(const uint4*)p;
            ((uint4*)&aq[kk])[1] = *(const uint4*)(p + 16);
        }
    }

    float run_m[8], run_l[8];
    f32x8 accv[4];
#pragma unroll
    for (int r = 0; r < 8; ++r) { run_m[r] = -1e30f; run_l[r] = 0.f; }
#pragma unroll
    for (int j = 0; j < 4; ++j)
#pragma unroll
        for (int r = 0; r < 8; ++r) accv[j][r] = 0.f;

    const int lr = tid >> 1;               // staging row 0..63
    const int lc = (tid & 1) * 32;         // 32-half chunk
    const unsigned short* gK = kn + (size_t)h * SEQ_S * EHEAD;
    const unsigned short* gV = vT + (size_t)h * EHEAD * SEQ_S;
    unsigned short* pbuf = smP + wid * 16 * KLD;

    for (int s0 = 0; s0 < SEQ_S; s0 += 64) {
        __syncthreads();                   // previous tile fully consumed
        {   // async-stage K tile (s-major 64x64) and V tile (e-major 64x64)
            const unsigned short* pk = gK + (size_t)(s0 + lr) * EHEAD + lc;
            const unsigned short* pv = gV + (size_t)lr * SEQ_S + s0 + lc;
#pragma unroll
            for (int q = 0; q < 4; ++q) {
                async_cp_b128(smK + lr * KLD + lc + 8 * q, pk + 8 * q);
                async_cp_b128(smV + lr * KLD + lc + 8 * q, pv + 8 * q);
            }
        }
        wait_async0();
        __syncthreads();

        // ---- scores: 16 x 64 (4 fragments), K-dim = E = 64 (2 wmma steps) ----
        f32x8 scf[4];
#pragma unroll
        for (int j = 0; j < 4; ++j)
#pragma unroll
            for (int r = 0; r < 8; ++r) scf[j][r] = 0.f;
#pragma unroll
        for (int kk = 0; kk < 2; ++kk)
#pragma unroll
            for (int j = 0; j < 4; ++j) {
                bf16x16 bk;
                const unsigned short* p = smK + (j * 16 + lid) * KLD + kk * 32 + hl * 8;
                ((uint4*)&bk)[0] = *(const uint4*)p;
                ((uint4*)&bk)[1] = *(const uint4*)(p + 16);
                scf[j] = __builtin_amdgcn_wmma_f32_16x16x32_bf16(
                    false, aq[kk], false, bk, (short)0, scf[j], false, false);
            }

        // ---- online softmax ----
        float nm[8], corr[8], psum[8];
#pragma unroll
        for (int r = 0; r < 8; ++r) {
            float m = scf[0][r];
            m = fmaxf(m, scf[1][r]); m = fmaxf(m, scf[2][r]); m = fmaxf(m, scf[3][r]);
            m *= scale;
            m = fmaxf(m, __shfl_xor(m, 1, 16)); m = fmaxf(m, __shfl_xor(m, 2, 16));
            m = fmaxf(m, __shfl_xor(m, 4, 16)); m = fmaxf(m, __shfl_xor(m, 8, 16));
            nm[r]   = fmaxf(run_m[r], m);
            corr[r] = __expf(run_m[r] - nm[r]);
            run_m[r] = nm[r];
            psum[r] = 0.f;
        }
#pragma unroll
        for (int j = 0; j < 4; ++j)
#pragma unroll
            for (int r = 0; r < 8; ++r) {
                float p = __expf(scf[j][r] * scale - nm[r]);
                psum[r] += p;
                pbuf[(r + 8 * hl) * KLD + j * 16 + lid] = f2bf(p);
            }
#pragma unroll
        for (int r = 0; r < 8; ++r) {
            float s = psum[r];
            s += __shfl_xor(s, 1, 16); s += __shfl_xor(s, 2, 16);
            s += __shfl_xor(s, 4, 16); s += __shfl_xor(s, 8, 16);
            run_l[r] = run_l[r] * corr[r] + s;
        }
#pragma unroll
        for (int j = 0; j < 4; ++j)
#pragma unroll
            for (int r = 0; r < 8; ++r) accv[j][r] *= corr[r];

        // ---- ctx += P (16 x 64) * V (64 x 64) ----
#pragma unroll
        for (int kk = 0; kk < 2; ++kk) {
            bf16x16 ap;
            const unsigned short* p = pbuf + lid * KLD + kk * 32 + hl * 8;
            ((uint4*)&ap)[0] = *(const uint4*)p;
            ((uint4*)&ap)[1] = *(const uint4*)(p + 16);
#pragma unroll
            for (int j = 0; j < 4; ++j) {
                bf16x16 bv;
                const unsigned short* q = smV + (j * 16 + lid) * KLD + kk * 32 + hl * 8;
                ((uint4*)&bv)[0] = *(const uint4*)q;
                ((uint4*)&bv)[1] = *(const uint4*)(q + 16);
                accv[j] = __builtin_amdgcn_wmma_f32_16x16x32_bf16(
                    false, ap, false, bv, (short)0, accv[j], false, false);
            }
        }
    }

    // ---- finalize: ctx /= running sum; bf16 row-major ----
#pragma unroll
    for (int r = 0; r < 8; ++r) run_l[r] = 1.f / run_l[r];
#pragma unroll
    for (int j = 0; j < 4; ++j) {
        const int gn = h * EHEAD + j * 16 + lid;
#pragma unroll
        for (int r = 0; r < 8; ++r) {
            const int gm = b * SEQ_S + lrow + r + 8 * hl;
            ctx[(size_t)gm * D_MODEL + gn] = f2bf(accv[j][r] * run_l[r]);
        }
    }
}

// ---------------------------------------------------------------------------
// Row LayerNorm: one block per row of 1024 fp32 values.
// ---------------------------------------------------------------------------
__global__ __launch_bounds__(256)
void layernorm_kernel(const float* __restrict__ x, const float* __restrict__ g,
                      const float* __restrict__ bta, float* __restrict__ out) {
    __shared__ float sbuf[16];
    __shared__ float mv[2];
    const int row = blockIdx.x, tid = threadIdx.x;
    const int lane = tid & 31, wid = tid >> 5;
    const float* xr = x + (size_t)row * D_MODEL;
    float4 v = *(const float4*)(xr + tid * 4);
    float s = v.x + v.y + v.z + v.w;
    float q = v.x * v.x + v.y * v.y + v.z * v.z + v.w * v.w;
#pragma unroll
    for (int off = 1; off < 32; off <<= 1) {
        s += __shfl_xor(s, off, 32);
        q += __shfl_xor(q, off, 32);
    }
    if (lane == 0) { sbuf[wid] = s; sbuf[8 + wid] = q; }
    __syncthreads();
    if (tid == 0) {
        float S = 0.f, Q = 0.f;
#pragma unroll
        for (int w = 0; w < 8; ++w) { S += sbuf[w]; Q += sbuf[8 + w]; }
        const float mu = S * (1.f / D_MODEL);
        mv[0] = mu;
        mv[1] = Q * (1.f / D_MODEL) - mu * mu;
    }
    __syncthreads();
    const float mu = mv[0];
    const float inv = rsqrtf(mv[1] + 1e-12f);
    float4 o;
    const int c = tid * 4;
    o.x = (v.x - mu) * inv * g[c + 0] + bta[c + 0];
    o.y = (v.y - mu) * inv * g[c + 1] + bta[c + 1];
    o.z = (v.z - mu) * inv * g[c + 2] + bta[c + 2];
    o.w = (v.w - mu) * inv * g[c + 3] + bta[c + 3];
    *(float4*)(out + (size_t)row * D_MODEL + c) = o;
}

// ---------------------------------------------------------------------------
// Host-side launch
// ---------------------------------------------------------------------------
extern "C" void kernel_launch(void* const* d_in, const int* in_sizes, int n_in,
                              void* d_out, int out_size, void* d_ws, size_t ws_size,
                              hipStream_t stream) {
    const float* tgt  = (const float*)d_in[0];
    const float* src  = (const float*)d_in[1];
    const float* val  = (const float*)d_in[2];
    const float* Wq   = (const float*)d_in[3];
    const float* bq   = (const float*)d_in[4];
    const float* Wk   = (const float*)d_in[5];
    const float* bk   = (const float*)d_in[6];
    const float* Wv   = (const float*)d_in[7];
    const float* bv   = (const float*)d_in[8];
    const float* Wo   = (const float*)d_in[9];
    const float* bo   = (const float*)d_in[10];
    const float* ln_g = (const float*)d_in[11];
    const float* ln_b = (const float*)d_in[12];

    char* ws = (char*)d_ws;
    const size_t MB = 1024 * 1024;
    unsigned short* WqT    = (unsigned short*)(ws + 0 * MB);
    unsigned short* WkT    = (unsigned short*)(ws + 2 * MB);
    unsigned short* WvT    = (unsigned short*)(ws + 4 * MB);
    unsigned short* WoT    = (unsigned short*)(ws + 6 * MB);
    unsigned short* tgt_bf = (unsigned short*)(ws + 8 * MB);
    unsigned short* src_bf = (unsigned short*)(ws + 16 * MB);
    unsigned short* val_bf = (unsigned short*)(ws + 20 * MB);
    unsigned short* qnb    = (unsigned short*)(ws + 24 * MB);
    unsigned short* knb    = (unsigned short*)(ws + 32 * MB);
    unsigned short* vTb    = (unsigned short*)(ws + 36 * MB);
    unsigned short* ctxb   = (unsigned short*)(ws + 40 * MB);
    float*          xbuf   = (float*)(ws + 48 * MB);

    cast_bf16_kernel<<<(ROWS_Q * D_MODEL) / 1024, 256, 0, stream>>>(tgt, tgt_bf, ROWS_Q * D_MODEL);
    cast_bf16_kernel<<<(SEQ_S * D_MODEL) / 1024, 256, 0, stream>>>(src, src_bf, SEQ_S * D_MODEL);
    cast_bf16_kernel<<<(SEQ_S * D_MODEL) / 1024, 256, 0, stream>>>(val, val_bf, SEQ_S * D_MODEL);

    dim3 tb(32, 8), tg(32, 32);
    transpose_cast_kernel<<<tg, tb, 0, stream>>>(Wq, WqT);
    transpose_cast_kernel<<<tg, tb, 0, stream>>>(Wk, WkT);
    transpose_cast_kernel<<<tg, tb, 0, stream>>>(Wv, WvT);
    transpose_cast_kernel<<<tg, tb, 0, stream>>>(Wo, WoT);

    gemm_wmma_kernel<<<dim3(ROWS_Q / 128, 8), 256, 0, stream>>>(tgt_bf, WqT, bq, nullptr, qnb, ROWS_Q, 0);
    gemm_wmma_kernel<<<dim3(SEQ_S / 128, 8), 256, 0, stream>>>(src_bf, WkT, bk, nullptr, knb, SEQ_S, 1);
    gemm_wmma_kernel<<<dim3(SEQ_S / 128, 8), 256, 0, stream>>>(val_bf, WvT, bv, nullptr, vTb, SEQ_S, 2);

    attn_wmma_kernel<<<dim3(SEQ_S / 64, NHEADS, 2), 128, 0, stream>>>(qnb, knb, vTb, ctxb);

    gemm_wmma_kernel<<<dim3(ROWS_Q / 128, 8), 256, 0, stream>>>(ctxb, WoT, bo, tgt, xbuf, ROWS_Q, 3);
    layernorm_kernel<<<ROWS_Q, 256, 0, stream>>>(xbuf, ln_g, ln_b, (float*)d_out);
}
